// ThymeLM_24206435680639
// MI455X (gfx1250) — matrix operand order repaired and checked by reading, not compiled, not run on hardware
//
#include <hip/hip_runtime.h>

#define VOCAB 32000
#define ED    512
#define NA    12
#define NC    24
#define NS    576   // NC*NC
#define HID   1024
#define NL    2
#define BB    8
#define TT    256
#define EPS   1e-5f

typedef __attribute__((ext_vector_type(16))) __bf16        v16bf;
typedef __attribute__((ext_vector_type(8)))  float         v8f;
typedef __attribute__((ext_vector_type(8)))  unsigned int  v8u;

__device__ __forceinline__ unsigned short f32_to_bf16_bits(float f) {
  unsigned u = __builtin_bit_cast(unsigned, f);
  unsigned r = u + 0x7FFFu + ((u >> 16) & 1u);
  return (unsigned short)(r >> 16);
}
__device__ __forceinline__ float bf16_bits_to_f32(unsigned short h) {
  unsigned u = ((unsigned)h) << 16;
  return __builtin_bit_cast(float, u);
}
__device__ __forceinline__ float gelu_exact(float x) {
  return 0.5f * x * (1.0f + erff(x * 0.70710678118654752f));
}

// ---------------------------------------------------------------------------
// Kernel 1: embed + (last layer only) tiny MLP -> c[B*T, 24]
// One block (128 threads) per token.
// ---------------------------------------------------------------------------
__global__ void token_mlp_kernel(const int* __restrict__ ids,
                                 const float* __restrict__ emb,
                                 const float* __restrict__ pos,
                                 const float* __restrict__ ax_w1,
                                 const float* __restrict__ ax_b1,
                                 const float* __restrict__ ax_g1,
                                 const float* __restrict__ ax_bt1,
                                 const float* __restrict__ ax_w2,
                                 const float* __restrict__ ax_b2,
                                 const float* __restrict__ co_w,
                                 const float* __restrict__ co_b,
                                 const float* __restrict__ co_g,
                                 const float* __restrict__ co_bt,
                                 float* __restrict__ c_out) {
  __shared__ float xs[ED];
  __shared__ float t1[2 * NA];
  __shared__ float t2[NA];
  __shared__ float mv[2];

  const int tok = blockIdx.x;
  const int t   = tok % TT;
  const int id  = ids[tok];
  const int L   = NL - 1;               // only last layer matters
  const int tid = threadIdx.x;

  const float* W1  = ax_w1  + (size_t)L * ED * (2 * NA);
  const float* B1  = ax_b1  + L * (2 * NA);
  const float* G1  = ax_g1  + L * (2 * NA);
  const float* BT1 = ax_bt1 + L * (2 * NA);
  const float* W2  = ax_w2  + (size_t)L * (2 * NA) * NA;
  const float* B2  = ax_b2  + L * NA;
  const float* CW  = co_w   + (size_t)L * NA * NC;
  const float* CB  = co_b   + L * NC;
  const float* CG  = co_g   + L * NC;
  const float* CBT = co_bt  + L * NC;

  for (int k = tid; k < ED; k += blockDim.x)
    xs[k] = emb[(size_t)id * ED + k] + pos[(size_t)t * ED + k];
  __syncthreads();

  if (tid < 2 * NA) {
    float s = B1[tid];
    for (int k = 0; k < ED; ++k) s += xs[k] * W1[k * (2 * NA) + tid];
    t1[tid] = s;
  }
  __syncthreads();
  if (tid == 0) {
    float m = 0.f, v = 0.f;
    for (int k = 0; k < 2 * NA; ++k) m += t1[k];
    m /= (float)(2 * NA);
    for (int k = 0; k < 2 * NA; ++k) { float d = t1[k] - m; v += d * d; }
    mv[0] = m; mv[1] = v / (float)(2 * NA);
  }
  __syncthreads();
  if (tid < 2 * NA) {
    float v = (t1[tid] - mv[0]) * rsqrtf(mv[1] + EPS) * G1[tid] + BT1[tid];
    t1[tid] = gelu_exact(v);
  }
  __syncthreads();
  if (tid < NA) {
    float s = B2[tid];
    for (int k = 0; k < 2 * NA; ++k) s += t1[k] * W2[k * NA + tid];
    t2[tid] = tanhf(s);
  }
  __syncthreads();
  if (tid < NC) {
    float s = CB[tid];
    for (int k = 0; k < NA; ++k) s += t2[k] * CW[k * NC + tid];
    t1[tid] = s;
  }
  __syncthreads();
  if (tid == 0) {
    float m = 0.f, v = 0.f;
    for (int k = 0; k < NC; ++k) m += t1[k];
    m /= (float)NC;
    for (int k = 0; k < NC; ++k) { float d = t1[k] - m; v += d * d; }
    mv[0] = m; mv[1] = v / (float)NC;
  }
  __syncthreads();
  if (tid < NC) {
    float v = (t1[tid] - mv[0]) * rsqrtf(mv[1] + EPS) * CG[tid] + CBT[tid];
    c_out[(size_t)tok * NC + tid] = tanhf(v);
  }
}

// ---------------------------------------------------------------------------
// Fragment-major plane layouts (bf16 hi + lo planes, ushort bits):
//   A-plane element (row, k):  mt=row/16, m=row%16, ks=k/32, kk=k%32,
//     halfA=(kk>>3)&1, i=(kk&7)+8*((kk>>4)&1), lane=m+16*halfA
//     index = ((mt*Ksteps + ks)*32 + lane)*16 + i
//   B-plane element (k, n):    ks=k/32, kk=k%32, halfB=(kk>>3)&1, e=(kk&7)+8*((kk>>4)&1),
//     nt=n/16, lane=(n&15)+16*halfB
//     index = ((ks*(N/16) + nt)*32 + lane)*16 + e
// Each lane's 16 bf16 fragment slice is 32B contiguous -> b128 loads.
// ---------------------------------------------------------------------------

// Kernel 2: decayed outer-product scan over T; writes states directly as
// fragment-major A hi/lo planes for GEMM1 (K=576 -> 18 K-steps).
__global__ void scan_pack_kernel(const float* __restrict__ c,
                                 const float* __restrict__ decay_raw,
                                 const float* __restrict__ mix_raw,
                                 unsigned short* __restrict__ Ahi,
                                 unsigned short* __restrict__ Alo,
                                 float* __restrict__ final_state) {
  const int b   = blockIdx.x;
  const int tid = threadIdx.x;          // 0..575 : this thread owns column k=tid
  const int i   = tid / NC;
  const int j   = tid % NC;
  const float decay = 1.f / (1.f + expf(-decay_raw[0]));
  const float mix   = 1.f / (1.f + expf(-mix_raw[0]));

  const int ks    = tid >> 5;
  const int kk    = tid & 31;
  const int halfA = (kk >> 3) & 1;
  const int ei    = (kk & 7) + (((kk >> 4) & 1) << 3);
  const int Ksteps = NS >> 5;           // 18

  const float* cb = c + (size_t)b * TT * NC;

  float st = 0.f;
  for (int t = 0; t < TT; ++t) {
    float ci = cb[t * NC + i];
    float cj = cb[t * NC + j];
    st = decay * st + mix * ci * cj;

    const int row  = b * TT + t;
    const int mt   = row >> 4;
    const int m    = row & 15;
    const int lane = m + 16 * halfA;
    const size_t o = (((size_t)mt * Ksteps + ks) * 32 + lane) * 16 + ei;
    unsigned short hb = f32_to_bf16_bits(st);
    Ahi[o] = hb;
    Alo[o] = f32_to_bf16_bits(st - bf16_bits_to_f32(hb));
  }
  final_state[(size_t)b * NS + tid] = st;
}

// One-time weight pack+swizzle: f32 [K,N] row-major -> fragment-major planes.
__global__ void pack_swizzle_B_kernel(const float* __restrict__ Bsrc,
                                      unsigned short* __restrict__ Bhi,
                                      unsigned short* __restrict__ Blo,
                                      int K, int N) {
  size_t idx = (size_t)blockIdx.x * blockDim.x + threadIdx.x;
  if (idx >= (size_t)K * N) return;
  int k = (int)(idx / N);
  int n = (int)(idx % N);
  float f = Bsrc[idx];

  int ks = k >> 5, kk = k & 31;
  int halfB = (kk >> 3) & 1, e = (kk & 7) + (((kk >> 4) & 1) << 3);
  int lane = (n & 15) + (halfB << 4);
  int nt = n >> 4;
  size_t o = (((size_t)ks * (N >> 4) + nt) * 32 + lane) * 16 + e;

  unsigned short hb = f32_to_bf16_bits(f);
  Bhi[o] = hb;
  Blo[o] = f32_to_bf16_bits(f - bf16_bits_to_f32(hb));
}

// ---------------------------------------------------------------------------
// Kernel 3: packed-plane WMMA GEMM, cache-reuse blocked.
// Block = 8 waves; wave w owns M-tile (bm*8+w) -> the 8 waves issue IDENTICAL
// B addresses (same 128-col group) so the WGP$ serves 7/8 B reads (L2 traffic
// for B drops ~8x). Each wave computes 16(M) x 128(N): 8 N-subtiles reuse one
// A fragment -> 24 x v_wmma_f32_16x16x32_bf16 per 32-wide K-step.
// mode 0: C[row*N+col] = acc + bias                (f32 row-major out)
// mode 1: pack gelu(acc+bias) into next GEMM's A planes (outKsteps K-steps)
// ---------------------------------------------------------------------------
__global__ void wmma_gemm_packed(const unsigned short* __restrict__ Ahi,
                                 const unsigned short* __restrict__ Alo,
                                 const unsigned short* __restrict__ Bhi,
                                 const unsigned short* __restrict__ Blo,
                                 const float* __restrict__ bias,
                                 float* __restrict__ Cout,
                                 unsigned short* __restrict__ Phi,
                                 unsigned short* __restrict__ Plo,
                                 int M, int N, int K, int mode, int outKsteps) {
  const int lane = threadIdx.x & 31;
  const int wv   = threadIdx.x >> 5;    // 0..7
  const int Ksteps = K >> 5;
  const int nBlk   = N >> 7;            // 128-wide column groups
  const int bm = blockIdx.x / nBlk;
  const int bn = blockIdx.x % nBlk;
  const int mt = bm * 8 + wv;           // this wave's 16-row tile
  if (mt * 16 >= M) return;             // wave-uniform exit

  const int nT16   = N >> 4;
  const int ntBase = bn << 3;           // first 16-col subtile of this block

  v8f acc[8] = {v8f{}, v8f{}, v8f{}, v8f{}, v8f{}, v8f{}, v8f{}, v8f{}};

  const v8u* pAhi = (const v8u*)Ahi + (size_t)mt * Ksteps * 32 + lane;
  const v8u* pAlo = (const v8u*)Alo + (size_t)mt * Ksteps * 32 + lane;
  const v8u* pBhi = (const v8u*)Bhi;
  const v8u* pBlo = (const v8u*)Blo;

  for (int ks = 0; ks < Ksteps; ++ks) {
    const v16bf a_hi = __builtin_bit_cast(v16bf, pAhi[(size_t)ks * 32]);
    const v16bf a_lo = __builtin_bit_cast(v16bf, pAlo[(size_t)ks * 32]);
    const size_t bbase = ((size_t)ks * nT16 + ntBase) * 32 + lane;
#pragma unroll
    for (int u = 0; u < 8; ++u) {
      const v16bf b_hi = __builtin_bit_cast(v16bf, pBhi[bbase + (size_t)u * 32]);
      const v16bf b_lo = __builtin_bit_cast(v16bf, pBlo[bbase + (size_t)u * 32]);
      acc[u] = __builtin_amdgcn_wmma_f32_16x16x32_bf16(false, a_hi, false, b_hi,
                                                       (short)0, acc[u], false, false);
      acc[u] = __builtin_amdgcn_wmma_f32_16x16x32_bf16(false, a_hi, false, b_lo,
                                                       (short)0, acc[u], false, false);
      acc[u] = __builtin_amdgcn_wmma_f32_16x16x32_bf16(false, a_lo, false, b_hi,
                                                       (short)0, acc[u], false, false);
    }
  }

  // C/D layout: VGPR v -> row = mt*16 + v + 8*(lane>>4), col = lane&15.
  const int m2base = (lane >> 4) << 3;
#pragma unroll
  for (int u = 0; u < 8; ++u) {
    const int col = (ntBase + u) * 16 + (lane & 15);
    const float bv = bias[col];
#pragma unroll
    for (int v = 0; v < 8; ++v) {
      const int row = mt * 16 + v + m2base;
      float r = acc[u][v] + bv;
      if (mode == 0) {
        Cout[(size_t)row * N + col] = r;
      } else {
        r = gelu_exact(r);
        // scatter into next GEMM's A-plane (row, k=col)
        const int ks2   = col >> 5;
        const int kk2   = col & 31;
        const int halfA = (kk2 >> 3) & 1;
        const int i2    = (kk2 & 7) + (((kk2 >> 4) & 1) << 3);
        const int lane2 = (v + m2base) + 16 * halfA;
        const size_t o  = (((size_t)mt * outKsteps + ks2) * 32 + lane2) * 16 + i2;
        unsigned short hb = f32_to_bf16_bits(r);
        Phi[o] = hb;
        Plo[o] = f32_to_bf16_bits(r - bf16_bits_to_f32(hb));
      }
    }
  }
}

// ---------------------------------------------------------------------------
extern "C" void kernel_launch(void* const* d_in, const int* in_sizes, int n_in,
                              void* d_out, int out_size, void* d_ws, size_t ws_size,
                              hipStream_t stream) {
  const int*   ids     = (const int*)d_in[0];
  const float* emb     = (const float*)d_in[1];
  const float* pos     = (const float*)d_in[2];
  const float* ax_w1   = (const float*)d_in[3];
  const float* ax_b1   = (const float*)d_in[4];
  const float* ax_g1   = (const float*)d_in[5];
  const float* ax_bt1  = (const float*)d_in[6];
  const float* ax_w2   = (const float*)d_in[7];
  const float* ax_b2   = (const float*)d_in[8];
  const float* co_w    = (const float*)d_in[9];
  const float* co_b    = (const float*)d_in[10];
  const float* co_g    = (const float*)d_in[11];
  const float* co_bt   = (const float*)d_in[12];
  const float* decay_r = (const float*)d_in[13];
  const float* mix_r   = (const float*)d_in[14];
  const float* out_w1  = (const float*)d_in[15];
  const float* out_b1  = (const float*)d_in[16];
  const float* out_w2  = (const float*)d_in[17];
  const float* out_b2  = (const float*)d_in[18];

  float* logits      = (float*)d_out;                              // [2048, 32000]
  float* final_state = (float*)d_out + (size_t)BB * TT * VOCAB;    // [8, 576]

  const int nTok = BB * TT;           // 2048
  const size_t szC   = (size_t)nTok * NC;                           // f32 count
  const size_t szA1  = (size_t)(nTok / 16) * (NS  / 32) * 32 * 16;  // ushort count
  const size_t szH   = (size_t)(nTok / 16) * (HID / 32) * 32 * 16;
  const size_t szW1  = (size_t)(NS  / 32) * (HID   / 16) * 32 * 16;
  const size_t szW2  = (size_t)(HID / 32) * (VOCAB / 16) * 32 * 16;

  unsigned char* w = (unsigned char*)d_ws;
  float*          c_ws = (float*)w;                 w += szC  * sizeof(float);
  unsigned short* A1hi = (unsigned short*)w;        w += szA1 * sizeof(unsigned short);
  unsigned short* A1lo = (unsigned short*)w;        w += szA1 * sizeof(unsigned short);
  unsigned short* Hhi  = (unsigned short*)w;        w += szH  * sizeof(unsigned short);
  unsigned short* Hlo  = (unsigned short*)w;        w += szH  * sizeof(unsigned short);
  unsigned short* W1hi = (unsigned short*)w;        w += szW1 * sizeof(unsigned short);
  unsigned short* W1lo = (unsigned short*)w;        w += szW1 * sizeof(unsigned short);
  unsigned short* W2hi = (unsigned short*)w;        w += szW2 * sizeof(unsigned short);
  unsigned short* W2lo = (unsigned short*)w;

  // 1) tokens -> c
  token_mlp_kernel<<<nTok, 128, 0, stream>>>(
      ids, emb, pos, ax_w1, ax_b1, ax_g1, ax_bt1, ax_w2, ax_b2,
      co_w, co_b, co_g, co_bt, c_ws);

  // 2) scan -> packed A1 planes + final_state
  scan_pack_kernel<<<BB, NS, 0, stream>>>(c_ws, decay_r, mix_r, A1hi, A1lo,
                                          final_state);

  // 3) one-time weight pack/swizzle
  {
    size_t n1 = (size_t)NS * HID;
    pack_swizzle_B_kernel<<<(unsigned)((n1 + 255) / 256), 256, 0, stream>>>(
        out_w1, W1hi, W1lo, NS, HID);
    size_t n2 = (size_t)HID * VOCAB;
    pack_swizzle_B_kernel<<<(unsigned)((n2 + 255) / 256), 256, 0, stream>>>(
        out_w2, W2hi, W2lo, HID, VOCAB);
  }

  // 4) GEMM1: h = gelu(states @ w1 + b1), packed epilogue -> H planes
  //    block tile 128x128 -> grid (2048/128)*(1024/128) = 16*8 = 128
  {
    int blocks = (nTok / 128) * (HID / 128);
    wmma_gemm_packed<<<blocks, 256, 0, stream>>>(
        A1hi, A1lo, W1hi, W1lo, out_b1,
        (float*)nullptr, Hhi, Hlo,
        nTok, HID, NS, 1, HID / 32);
  }
  // 5) GEMM2: logits = h @ w2 + b2
  //    block tile 128x128 -> grid (2048/128)*(32000/128) = 16*250 = 4000
  {
    int blocks = (nTok / 128) * (VOCAB / 128);
    wmma_gemm_packed<<<blocks, 256, 0, stream>>>(
        Hhi, Hlo, W2hi, W2lo, out_b2,
        logits, (unsigned short*)nullptr, (unsigned short*)nullptr,
        nTok, VOCAB, HID, 0, 0);
  }
}
